// KrabbyPatty_25649544692323
// MI455X (gfx1250) — matrix operand order repaired and checked by reading, not compiled, not run on hardware
//
#include <hip/hip_runtime.h>
#include <utility>

// Problem constants (match reference)
#define LSEQ 4096
#define BB   8
#define DM   1024
#define RR   32
#define EPSF 1e-9f
#define KSTEPS 6

typedef __bf16 bf16;
typedef __attribute__((ext_vector_type(16))) __bf16 bf16x16;
typedef __attribute__((ext_vector_type(8)))  __bf16 bf16x8;
typedef __attribute__((ext_vector_type(8)))  float  f32x8;
typedef __attribute__((ext_vector_type(4)))  float  f32x4;

// ---- WMMA operand loaders -------------------------------------------------
// A (16x32 bf16): lane m = lane%16. lanes<16 hold K {0..7,16..23}, lanes>=16
// hold K {8..15,24..31}  -> two contiguous b128 loads per lane.
__device__ __forceinline__ bf16x16 load_a16(const bf16* __restrict__ rowp,
                                            int kblk, int lane) {
  const int ko = kblk + ((lane >> 4) << 3);
  bf16x8 lo = *(const bf16x8*)(rowp + ko);
  bf16x8 hi = *(const bf16x8*)(rowp + ko + 16);
  bf16x16 v;
#pragma unroll
  for (int i = 0; i < 8; ++i) { v[i] = lo[i]; v[i + 8] = hi[i]; }
  return v;
}
// B (32x16 bf16) loaded from B^T (row n, contiguous k): lanes<16 hold K {0..7,16..23},
// lanes>=16 hold K {8..15,24..31} -> two contiguous b128 loads per lane.
__device__ __forceinline__ bf16x16 load_b16(const bf16* __restrict__ rowp,
                                            int kblk, int lane) {
  const int ko = kblk + ((lane >> 4) << 3);
  bf16x8 lo = *(const bf16x8*)(rowp + ko);
  bf16x8 hi = *(const bf16x8*)(rowp + ko + 16);
  bf16x16 v;
#pragma unroll
  for (int i = 0; i < 8; ++i) { v[i] = lo[i]; v[i + 8] = hi[i]; }
  return v;
}

__device__ __forceinline__ f32x8 wmma_bf16(bf16x16 a, bf16x16 b, f32x8 c) {
  return __builtin_amdgcn_wmma_f32_16x16x32_bf16(false, a, false, b,
                                                 (short)0, c, false, false);
}

// Quad of B tiles (B^T rows n, n+16, n+32, n+48) for one K-chunk.
// All 8 b128 loads issue as one clause BEFORE the 4-WMMA burst, so the
// waits preceding the WMMAs are partial (descending loadcnt), not drains.
struct B4 { bf16x16 b0, b1, b2, b3; };

__device__ __forceinline__ B4 load_b4(const bf16* __restrict__ brow, int ldb,
                                      int k, int lane) {
  B4 r;
  r.b0 = load_b16(brow,                    k, lane);
  r.b1 = load_b16(brow + 16 * (size_t)ldb, k, lane);
  r.b2 = load_b16(brow + 32 * (size_t)ldb, k, lane);
  r.b3 = load_b16(brow + 48 * (size_t)ldb, k, lane);
  return r;
}

__device__ __forceinline__ void wmma4_acc(bf16x16 a, const B4& b, f32x8& c0,
                                          f32x8& c1, f32x8& c2, f32x8& c3) {
  c0 = wmma_bf16(a, b.b0, c0);
  c1 = wmma_bf16(a, b.b1, c1);
  c2 = wmma_bf16(a, b.b2, c2);
  c3 = wmma_bf16(a, b.b3, c3);
}

// 16x64 strip: ~80 live VGPRs (32 acc + 8 A + 32 B) -> no spills, >=2
// waves/SIMD hide remaining latency across chunks.
__device__ __forceinline__ void gemm_strip4(const bf16* __restrict__ A, int lda,
                                            const bf16* __restrict__ Bt, int ldb,
                                            int mbase, int n0, int k0, int klen,
                                            int lane, f32x8& c0, f32x8& c1,
                                            f32x8& c2, f32x8& c3) {
  const int nlo = lane & 15;
  const bf16* arow = A  + (size_t)(mbase + nlo) * lda;
  const bf16* brow = Bt + (size_t)(n0 + nlo) * ldb;
  for (int k = k0; k < k0 + klen; k += 32) {
    bf16x16 a = load_a16(arow, k, lane);
    B4      b = load_b4(brow, ldb, k, lane);
    wmma4_acc(a, b, c0, c1, c2, c3);
  }
}

// Single-tile GEMM for the tiny Gram / C@W2 reductions.
__device__ __forceinline__ f32x8 gemm_tile(const bf16* __restrict__ arow,
                                           const bf16* __restrict__ brow,
                                           int k0, int klen, int lane) {
  f32x8 acc = {};
  for (int k = k0; k < k0 + klen; k += 32)
    acc = wmma_bf16(load_a16(arow, k, lane), load_b16(brow, k, lane), acc);
  return acc;
}

// Transposed f32 store of one 16x16 tile: O[n][m], contiguous in m per lane.
__device__ __forceinline__ void store_tr(float* __restrict__ O, int ldo,
                                         int mbase, int n, int lane, f32x8 acc) {
  const int hi = lane >> 4;
  float* p = O + (size_t)n * ldo + mbase + 8 * hi;
  f32x4 v0 = {acc[0], acc[1], acc[2], acc[3]};
  f32x4 v1 = {acc[4], acc[5], acc[6], acc[7]};
  *(f32x4*)p = v0; *(f32x4*)(p + 4) = v1;
}

// ---- Prep kernels ---------------------------------------------------------
__global__ void k_wtrans(const float* __restrict__ W, bf16* __restrict__ Wt) {
  const int i = blockIdx.x * blockDim.x + threadIdx.x;
  const int n = i & (DM - 1), k = i >> 10;
  Wt[(size_t)n * DM + k] = (bf16)W[(size_t)k * DM + n];
}

__global__ void k_initD(const float* __restrict__ Dinit, float* __restrict__ D1,
                        bf16* __restrict__ Dbf, bf16* __restrict__ Dtbf) {
  const int i = blockIdx.x * blockDim.x + threadIdx.x;   // B*L*R
  const int b = i >> 17;
  const int l = (i >> 5) & (LSEQ - 1);
  const int r = i & (RR - 1);
  const float v = Dinit[l * RR + r];
  D1[i]  = v;
  Dbf[i] = (bf16)v;
  Dtbf[((size_t)b * RR + r) * LSEQ + l] = (bf16)v;
}

__global__ void k_initC(const float* __restrict__ Cinit, float* __restrict__ C1,
                        bf16* __restrict__ Cbf) {
  const int i = blockIdx.x * blockDim.x + threadIdx.x;   // B*R*D
  const float v = Cinit[i & (RR * DM - 1)];
  C1[i]  = v;
  Cbf[i] = (bf16)v;
}

// ---- GEMM1: Y = relu(X(l,b,:)@W1 + b1), 16x64/wave, dual-layout output ----
__global__ void k_gemm1(const float* __restrict__ X, const bf16* __restrict__ W1t,
                        const float* __restrict__ b1, bf16* __restrict__ Ybf,
                        bf16* __restrict__ Ytbf) {
  const int lane = threadIdx.x & 31, wave = threadIdx.x >> 5;
  const int nlo = lane & 15, hi = lane >> 4;
  const int mbase = blockIdx.y * 64 + (wave & 3) * 16;   // m = b*L + l
  const int n0    = blockIdx.x * 256 + (wave >> 2) * 64;
  const int m_lane = mbase + nlo;
  const int b = m_lane >> 12, l = m_lane & (LSEQ - 1);
  const float* arow = X + ((size_t)l * BB + b) * DM;     // contiguous in d
  const bf16*  brow = W1t + (size_t)(n0 + nlo) * DM;
  const int ko0 = hi << 3;
  f32x8 c0 = {}, c1 = {}, c2 = {}, c3 = {};
  for (int kblk = 0; kblk < DM; kblk += 32) {
    __builtin_prefetch(arow + kblk + 64, 0, 3);          // HBM stream: X
    const int ko = kblk + ko0;
    f32x4 a0 = *(const f32x4*)(arow + ko);
    f32x4 a1 = *(const f32x4*)(arow + ko + 4);
    f32x4 a2 = *(const f32x4*)(arow + ko + 16);
    f32x4 a3 = *(const f32x4*)(arow + ko + 20);
    B4 bq = load_b4(brow, DM, kblk, lane);
    bf16x16 a;
#pragma unroll
    for (int i = 0; i < 4; ++i) {
      a[i] = (bf16)a0[i]; a[i + 4] = (bf16)a1[i];
      a[i + 8] = (bf16)a2[i]; a[i + 12] = (bf16)a3[i];
    }
    wmma4_acc(a, bq, c0, c1, c2, c3);
  }

  const int bt = mbase >> 12;
  const int lt = (mbase & (LSEQ - 1)) + 8 * hi;
  f32x8 accs[4] = {c0, c1, c2, c3};
#pragma unroll
  for (int t = 0; t < 4; ++t) {
    const int n = n0 + nlo + 16 * t;
    const float bv = b1[n];
    bf16x8 yt;
#pragma unroll
    for (int j = 0; j < 8; ++j) {
      float y = accs[t][j] + bv;
      y = y > 0.f ? y : 0.f;
      const bf16 yb = (bf16)y;
      Ybf[(size_t)(mbase + 8 * hi + j) * DM + n] = yb;
      yt[j] = yb;
    }
    *(bf16x8*)(Ytbf + ((size_t)bt * DM + n) * LSEQ + lt) = yt;  // Yt[b][d][l]
  }
}

// ---- DtX partials: P[g][b][d][r] = (D^T X)[r][d] over K-slice g -----------
__global__ void k_dtx(const bf16* __restrict__ Dtbf, const bf16* __restrict__ Ytbf,
                      float* __restrict__ P) {
  const int lane = threadIdx.x & 31, wave = threadIdx.x >> 5;
  const int nlo = lane & 15;
  const int mbase = (wave & 1) * 16;                      // r
  const int n0 = blockIdx.x * 256 + (wave >> 1) * 64;     // d
  const int g = blockIdx.y, b = blockIdx.z;
  f32x8 c0 = {}, c1 = {}, c2 = {}, c3 = {};
  gemm_strip4(Dtbf + (size_t)b * RR * LSEQ, LSEQ,
              Ytbf + (size_t)b * DM * LSEQ, LSEQ,
              mbase, n0, g * 1024, 1024, lane, c0, c1, c2, c3);
  float* base = P + ((size_t)g * BB + b) * DM * RR;
  store_tr(base, RR, mbase, n0 + nlo,      lane, c0);
  store_tr(base, RR, mbase, n0 + nlo + 16, lane, c1);
  store_tr(base, RR, mbase, n0 + nlo + 32, lane, c2);
  store_tr(base, RR, mbase, n0 + nlo + 48, lane, c3);
}

// ---- Gram partials: P[g][b][m][n] = sum_{k in slice} M[m][k]M[n][k] -------
__global__ void k_gram(const bf16* __restrict__ M, int ld, int kslice,
                       float* __restrict__ P) {
  const int lane = threadIdx.x & 31, wave = threadIdx.x >> 5;
  const int mbase = (wave & 1) * 16, nbase = (wave >> 1) * 16;
  const int g = blockIdx.x, b = blockIdx.y;
  const bf16* base = M + (size_t)b * RR * ld;
  const int nlo = lane & 15, hi = lane >> 4;
  f32x8 acc = gemm_tile(base + (size_t)(mbase + nlo) * ld,
                        base + (size_t)(nbase + nlo) * ld,
                        g * kslice, kslice, lane);
#pragma unroll
  for (int j = 0; j < 8; ++j)
    P[((size_t)g * BB + b) * RR * RR + (mbase + 8 * hi + j) * RR + nbase + nlo] = acc[j];
}

__global__ void k_gram_red(const float* __restrict__ P, float* __restrict__ Out,
                           int G) {
  const int i = blockIdx.x * blockDim.x + threadIdx.x;    // B*R*R
  float s = 0.f;
  for (int g = 0; g < G; ++g) s += P[(size_t)g * (BB * RR * RR) + i];
  Out[i] = s;
}

// ---- C <- C * DtX / (DtD@C + eps); sums the 4 DtX K-partials --------------
__global__ void k_cupdate(const float* __restrict__ Csrc, const float* __restrict__ DtD,
                          const float* __restrict__ DtXP, float* __restrict__ Cdst,
                          bf16* __restrict__ Cbf) {
  const int i = blockIdx.x * blockDim.x + threadIdx.x;   // B*R*D
  const int b = i >> 15, r = (i >> 10) & (RR - 1), d = i & (DM - 1);
  const float* dtd = DtD + b * (RR * RR) + r * RR;
  const float* c   = Csrc + (size_t)b * RR * DM;
  float den = EPSF;
#pragma unroll
  for (int q = 0; q < RR; ++q) den += dtd[q] * c[q * DM + d];
  float num = 0.f;
#pragma unroll
  for (int g = 0; g < 4; ++g)
    num += DtXP[(((size_t)g * BB + b) * DM + d) * RR + r];
  const float v = c[r * DM + d] * num / den;
  Cdst[(size_t)b * RR * DM + r * DM + d] = v;
  Cbf [(size_t)b * RR * DM + r * DM + d] = (bf16)v;
}

// ---- XCt[b][l][r] via (C @ Y^T)[r][l], 16x64 strips  (M=R,N=L,K=D) --------
__global__ void k_xct(const bf16* __restrict__ Cbf, const bf16* __restrict__ Ybf,
                      float* __restrict__ XCt) {
  const int lane = threadIdx.x & 31, wave = threadIdx.x >> 5;
  const int nlo = lane & 15;
  const int mbase = (wave & 1) * 16;                      // r
  const int n0 = blockIdx.x * 256 + (wave >> 1) * 64;     // l
  const int b = blockIdx.y;
  f32x8 c0 = {}, c1 = {}, c2 = {}, c3 = {};
  gemm_strip4(Cbf + (size_t)b * RR * DM, DM,
              Ybf + (size_t)b * LSEQ * DM, DM,
              mbase, n0, 0, DM, lane, c0, c1, c2, c3);
  float* base = XCt + (size_t)b * LSEQ * RR;
  store_tr(base, RR, mbase, n0 + nlo,      lane, c0);
  store_tr(base, RR, mbase, n0 + nlo + 16, lane, c1);
  store_tr(base, RR, mbase, n0 + nlo + 32, lane, c2);
  store_tr(base, RR, mbase, n0 + nlo + 48, lane, c3);
}

// ---- D <- D * XCt / (D@CCt + eps), ping-pong + dual bf16 layouts ----------
__global__ void k_dupdate(const float* __restrict__ Dsrc, const float* __restrict__ CCt,
                          const float* __restrict__ XCt, float* __restrict__ Ddst,
                          bf16* __restrict__ Dbf, bf16* __restrict__ Dtbf) {
  const int i = blockIdx.x * blockDim.x + threadIdx.x;   // B*L*R
  const int b = i >> 17, l = (i >> 5) & (LSEQ - 1), r = i & (RR - 1);
  const float* drow = Dsrc + ((size_t)b * LSEQ + l) * RR;
  const float* cct  = CCt + b * (RR * RR);
  float den = EPSF;
#pragma unroll
  for (int q = 0; q < RR; ++q) den += drow[q] * cct[q * RR + r];
  const float v = drow[r] * XCt[((size_t)b * LSEQ + l) * RR + r] / den;
  Ddst[((size_t)b * LSEQ + l) * RR + r] = v;
  Dbf [((size_t)b * LSEQ + l) * RR + r] = (bf16)v;
  Dtbf[((size_t)b * RR + r) * LSEQ + l] = (bf16)v;
}

// ---- CW2^T[b][d][r] = (C@W2)[r][d]  (M=D, N=R, K=D) -----------------------
__global__ void k_cw2t(const bf16* __restrict__ W2t, const bf16* __restrict__ Cbf,
                       bf16* __restrict__ CW2t) {
  const int lane = threadIdx.x & 31, wave = threadIdx.x >> 5;
  const int nbase = wave * 16;        // r (N=32, 2 waves)
  const int mbase = blockIdx.x * 16;  // d
  const int b = blockIdx.y;
  const int nlo = lane & 15, hi = lane >> 4;
  f32x8 acc = gemm_tile(W2t + (size_t)(mbase + nlo) * DM,
                        Cbf + (size_t)b * RR * DM + (size_t)(nbase + nlo) * DM,
                        0, DM, lane);
#pragma unroll
  for (int j = 0; j < 8; ++j)
    CW2t[(size_t)b * DM * RR + (mbase + 8 * hi + j) * RR + nbase + nlo] = (bf16)acc[j];
}

// ---- out(l,b,d) = D @ CW2 + b2 : K=32, one A fetch -> 4 WMMAs -------------
__global__ void k_final(const bf16* __restrict__ Dbf, const bf16* __restrict__ CW2t,
                        const float* __restrict__ b2, float* __restrict__ Out) {
  const int lane = threadIdx.x & 31, wave = threadIdx.x >> 5;
  const int nlo = lane & 15, hi = lane >> 4;
  const int mbase = blockIdx.y * 64 + (wave & 3) * 16;   // m = b*L + l
  const int n0    = blockIdx.x * 256 + (wave >> 2) * 64; // d
  const int b = mbase >> 12;
  const bf16* arow = Dbf + (size_t)(mbase + nlo) * RR;
  const bf16* brow = CW2t + ((size_t)b * DM + n0 + nlo) * RR;
  f32x8 c0 = {}, c1 = {}, c2 = {}, c3 = {};
  bf16x16 a = load_a16(arow, 0, lane);
  B4 bq = load_b4(brow, RR, 0, lane);
  wmma4_acc(a, bq, c0, c1, c2, c3);
  f32x8 accs[4] = {c0, c1, c2, c3};
#pragma unroll
  for (int t = 0; t < 4; ++t) {
    const int n = n0 + nlo + 16 * t;
    const float bv = b2[n];
#pragma unroll
    for (int j = 0; j < 8; ++j) {
      const int mg = mbase + 8 * hi + j;
      const int l = mg & (LSEQ - 1);
      Out[((size_t)l * BB + b) * DM + n] = accs[t][j] + bv;
    }
  }
}

// ---- host -----------------------------------------------------------------
extern "C" void kernel_launch(void* const* d_in, const int* in_sizes, int n_in,
                              void* d_out, int out_size, void* d_ws, size_t ws_size,
                              hipStream_t stream) {
  (void)in_sizes; (void)n_in; (void)out_size; (void)ws_size;
  const float* X     = (const float*)d_in[0];
  const float* Dinit = (const float*)d_in[1];
  const float* Cinit = (const float*)d_in[2];
  const float* W1    = (const float*)d_in[3];
  const float* b1    = (const float*)d_in[4];
  const float* W2    = (const float*)d_in[5];
  const float* b2    = (const float*)d_in[6];
  float* Out = (float*)d_out;

  char* ws = (char*)d_ws;
  size_t off = 0;
  auto alloc = [&](size_t bytes) -> void* {
    void* p = ws + off;
    off = (off + bytes + 255) & ~(size_t)255;
    return p;
  };
  bf16*  Ybf   = (bf16*) alloc((size_t)BB * LSEQ * DM * 2);   // 64 MB, L2-resident
  bf16*  Ytbf  = (bf16*) alloc((size_t)BB * DM * LSEQ * 2);   // 64 MB
  bf16*  W1t   = (bf16*) alloc((size_t)DM * DM * 2);
  bf16*  W2t   = (bf16*) alloc((size_t)DM * DM * 2);
  float* D1    = (float*)alloc((size_t)BB * LSEQ * RR * 4);
  float* D2    = (float*)alloc((size_t)BB * LSEQ * RR * 4);
  bf16*  Dbf   = (bf16*) alloc((size_t)BB * LSEQ * RR * 2);
  bf16*  Dtbf  = (bf16*) alloc((size_t)BB * RR * LSEQ * 2);
  float* C1    = (float*)alloc((size_t)BB * RR * DM * 4);
  float* C2    = (float*)alloc((size_t)BB * RR * DM * 4);
  bf16*  Cbf   = (bf16*) alloc((size_t)BB * RR * DM * 2);
  float* DtXP  = (float*)alloc((size_t)4 * BB * DM * RR * 4);   // 4 K-partials
  float* XCt   = (float*)alloc((size_t)BB * LSEQ * RR * 4);
  float* DtDP  = (float*)alloc((size_t)8 * BB * RR * RR * 4);   // 8 K-partials
  float* CCtP  = (float*)alloc((size_t)4 * BB * RR * RR * 4);   // 4 K-partials
  float* DtD   = (float*)alloc((size_t)BB * RR * RR * 4);
  float* CCt   = (float*)alloc((size_t)BB * RR * RR * 4);
  bf16*  CW2t  = (bf16*) alloc((size_t)BB * DM * RR * 2);

  k_wtrans<<<4096, 256, 0, stream>>>(W1, W1t);
  k_wtrans<<<4096, 256, 0, stream>>>(W2, W2t);
  k_initD <<<4096, 256, 0, stream>>>(Dinit, D1, Dbf, Dtbf);
  k_initC <<<1024, 256, 0, stream>>>(Cinit, C1, Cbf);
  k_gemm1 <<<dim3(4, 512), 256, 0, stream>>>(X, W1t, b1, Ybf, Ytbf);

  float *Ds = D1, *Dd = D2, *Cs = C1, *Cd = C2;
  for (int s = 0; s < KSTEPS; ++s) {
    k_dtx     <<<dim3(4, 4, 8), 256, 0, stream>>>(Dtbf, Ytbf, DtXP);
    k_gram    <<<dim3(8, 8), 128, 0, stream>>>(Dtbf, LSEQ, 512, DtDP);
    k_gram_red<<<32, 256, 0, stream>>>(DtDP, DtD, 8);
    k_cupdate <<<1024, 256, 0, stream>>>(Cs, DtD, DtXP, Cd, Cbf);
    k_xct     <<<dim3(16, 8), 256, 0, stream>>>(Cbf, Ybf, XCt);
    k_gram    <<<dim3(4, 8), 128, 0, stream>>>(Cbf, DM, 256, CCtP);
    k_gram_red<<<32, 256, 0, stream>>>(CCtP, CCt, 4);
    k_dupdate <<<4096, 256, 0, stream>>>(Ds, CCt, XCt, Dd, Dbf, Dtbf);
    std::swap(Ds, Dd);
    std::swap(Cs, Cd);
  }

  k_cw2t <<<dim3(64, 8), 64, 0, stream>>>(W2t, Cbf, CW2t);
  k_final<<<dim3(4, 512), 256, 0, stream>>>(Dbf, CW2t, b2, Out);
}
